// MemModule_46712064311885
// MI455X (gfx1250) — compile-verified, matrix operand order checked by reading, not run on hardware
//
#include <hip/hip_runtime.h>
#include <hip/hip_bf16.h>

typedef __attribute__((ext_vector_type(16))) __bf16 v16bf;
typedef __attribute__((ext_vector_type(8)))  float  v8f;
typedef __attribute__((ext_vector_type(4)))  unsigned int u32x4;
typedef __attribute__((ext_vector_type(8)))  int i32x8;
typedef __attribute__((ext_vector_type(4)))  int i32x4;

#define BDIM   256
#define ROWS   16
#define C_DIM  256
#define M_DIM  2000
#define M_PAD  2016
#define HW     1024          // H*W = 32*32
#define N_TOT  32768         // B*H*W
#define LAMBDA 0.0025f
#define EPSV   1e-12f

#define W1_ELEMS (M_DIM * C_DIM)   // bf16 weight, row-major [M][C]  (GEMM1 B)
#define W2_ELEMS (C_DIM * M_PAD)   // bf16 weight, transposed [C][M_PAD] (GEMM2 B)

#if __has_builtin(__builtin_amdgcn_tensor_load_to_lds) && \
    __has_builtin(__builtin_amdgcn_s_wait_tensorcnt)
#define USE_TDM 1
#endif

// ---------------------------------------------------------------------------
// Prep: f32 weight -> bf16 copies in workspace (row-major + transposed/padded)
// ---------------------------------------------------------------------------
__global__ void prep_weights(const float* __restrict__ w,
                             __bf16* __restrict__ w1,
                             __bf16* __restrict__ w2) {
  int t = blockIdx.x * blockDim.x + threadIdx.x;
  if (t < W1_ELEMS) {
    int m = t / C_DIM, c = t % C_DIM;
    float f = w[t];
    w1[t] = (__bf16)f;
    w2[c * M_PAD + m] = (__bf16)f;
  }
  // zero the K-padding columns of the transposed copy (m = 2000..2015)
  if (t < C_DIM * (M_PAD - M_DIM)) {
    int c = t / (M_PAD - M_DIM);
    int mp = M_DIM + (t % (M_PAD - M_DIM));
    w2[c * M_PAD + mp] = (__bf16)0.0f;
  }
}

// ---------------------------------------------------------------------------
// Fused: GEMM1 -> softmax/hardshrink/L1 -> GEMM2, 16 N-rows per workgroup
// ---------------------------------------------------------------------------
__global__ __launch_bounds__(BDIM)
void memae_fused(const float*  __restrict__ x,
                 const __bf16* __restrict__ w1,
                 const __bf16* __restrict__ w2,
                 float* __restrict__ outY,
                 float* __restrict__ outAtt,
                 float* __restrict__ outPre) {
  // Phase 1-2: xStage f32 [256][16] @0 (16 KB), zA bf16 [16][256] @16 KB.
  // Phase 3-4: attB bf16 [16][2016] @0 (64512 B). Barrier-separated aliasing.
  __shared__ __align__(16) unsigned char smem[ROWS * M_PAD * 2];
  float*  xStage = (float*)smem;                    // [256][16] f32
  __bf16* zA     = (__bf16*)(smem + 16384);         // [16][256] bf16
  __bf16* attB   = (__bf16*)smem;                   // [16][2016] bf16

  const int tid  = threadIdx.x;
  const int lane = tid & 31;
  const int wave = tid >> 5;
  const int n0   = blockIdx.x * ROWS;   // first row of this tile
  const int b    = n0 >> 10;            // batch image (HW = 1024)
  const int hw0  = n0 & 1023;           // 16-aligned offset inside image

  // ---- Phase 1: stage z-tile (16 rows x 256 ch) as bf16 in LDS -----------
#ifdef USE_TDM
  // Tensor Data Mover: DMA the 2D tile (16 contiguous f32 x 256 lines,
  // line stride HW) from global x into LDS. Issued by wave 0 only.
  if (wave == 0) {
    const unsigned long long ga =
        (unsigned long long)(uintptr_t)(x + (size_t)b * C_DIM * HW + hw0);
    const unsigned int lds_off = (unsigned int)(uintptr_t)xStage; // LDS aperture: low 32b

    u32x4 g0;
    g0.x = 1u;                                   // count=1, user descriptor
    g0.y = lds_off;                              // lds_addr
    g0.z = (unsigned int)(ga & 0xFFFFFFFFu);     // global_addr[31:0]
    g0.w = ((unsigned int)(ga >> 32) & 0x01FFFFFFu) | (2u << 30);  // [56:32] | type=2

    i32x8 g1;
    g1[0] = (int)(2u << 16);        // data_size = 4 bytes
    g1[1] = (int)(HW << 16);        // tensor_dim0[15:0]   (bits 63:48)
    g1[2] = (int)(C_DIM << 16);     // tensor_dim1[15:0]   (bits 95:80)
    g1[3] = (int)(ROWS << 16);      // tile_dim0 = 16      (bits 127:112)
    g1[4] = C_DIM;                  // tile_dim1 = 256     (bits 143:128)
    g1[5] = HW;                     // tensor_dim0_stride  (bits 191:160)
    g1[6] = 0;
    g1[7] = 0;

    i32x4 gz = {0, 0, 0, 0};
#if __has_include(<hip/amd_detail/amd_gfx1250_TDM.h>)
    i32x8 gz8 = {0, 0, 0, 0, 0, 0, 0, 0};
    __builtin_amdgcn_tensor_load_to_lds(g0, g1, gz, gz, gz8, 0);   // clang-23 6-arg
#else
    __builtin_amdgcn_tensor_load_to_lds(g0, g1, gz, gz, 0);        // ROCm 7.2 5-arg
#endif
    __builtin_amdgcn_s_wait_tensorcnt(0);
  }
  __syncthreads();
  {
    const int c = tid;  // 256 threads == 256 channels
    #pragma unroll 4
    for (int r = 0; r < ROWS; ++r)
      zA[r * C_DIM + c] = (__bf16)xStage[c * ROWS + r];
  }
#else
  {
    const float* xp = x + (size_t)b * C_DIM * HW + hw0;  // + c*HW + r
    const int c = tid;
    #pragma unroll 4
    for (int r = 0; r < ROWS; ++r)
      zA[r * C_DIM + c] = (__bf16)xp[(size_t)c * HW + r];
  }
#endif
  __syncthreads();

  const int arow  = lane & 15;
  const int khalf = (lane >> 4) * 16;

  // Hoist all 8 A-fragments (K = 256) — shared by every M-tile of GEMM1.
  v16bf afrag[8];
  #pragma unroll
  for (int kk = 0; kk < 8; ++kk)
    afrag[kk] = *(const v16bf*)(zA + arow * C_DIM + kk * 32 + khalf);

  // ---- Phase 2: GEMM1  pre = z @ W^T  (125 M-tiles split across 8 waves) --
  for (int mt = wave; mt < 125; mt += 8) {
    const __bf16* bp = w1 + (size_t)(mt * 16 + arow) * C_DIM + khalf;
    // branch-free prefetch of the next tile's B rows (clamped, wave-uniform)
    const int mtn = (mt + 8 < 125) ? (mt + 8) : mt;
    __builtin_prefetch(w1 + (size_t)(mtn * 16 + arow) * C_DIM, 0, 1);

    v16bf bfrag[8];
    #pragma unroll
    for (int kk = 0; kk < 8; ++kk)
      bfrag[kk] = *(const v16bf*)(bp + kk * 32);

    v8f acc = {};
    #pragma unroll
    for (int kk = 0; kk < 8; ++kk)
      acc = __builtin_amdgcn_wmma_f32_16x16x32_bf16(
          false, afrag[kk], false, bfrag[kk], (short)0, acc, false, false);

    // D layout: lane -> m column (lane&15), VGPR v -> n row v + 8*(lane>>4)
    size_t p = (size_t)b * M_DIM * HW + (size_t)(mt * 16 + arow) * HW
             + (size_t)hw0 + 8 * (lane >> 4);
    #pragma unroll
    for (int v = 0; v < 8; ++v) outPre[p + v] = acc[v];

#if __has_builtin(__builtin_amdgcn_sched_group_barrier)
    // Pin schedule: all 16 B-loads first, then the 8-WMMA chain, then the
    // 2 result stores -> partial loadcnt waits, loads overlap WMMAs.
    __builtin_amdgcn_sched_group_barrier(0x020, 16, 0); // VMEM read x16
    __builtin_amdgcn_sched_group_barrier(0x008, 8, 0);  // WMMA x8
    __builtin_amdgcn_sched_group_barrier(0x040, 2, 0);  // VMEM write x2
#endif
  }

  __threadfence();     // make pre visible to sibling waves (L2/WGP$)
  __syncthreads();

  // ---- Phase 3: softmax + hard-shrink-relu + L1 norm ----------------------
  // 16 threads per row, both half-rows of a wave => width-16 shuffles.
  {
    const int r = tid >> 4;
    const int j = tid & 15;
    const float* preRow = outPre + (size_t)b * M_DIM * HW + hw0 + r;  // + m*HW
    float mx = -3.4e38f;
    for (int m = j; m < M_DIM; m += 16)
      mx = fmaxf(mx, preRow[(size_t)m * HW]);
    #pragma unroll
    for (int off = 8; off > 0; off >>= 1)
      mx = fmaxf(mx, __shfl_xor(mx, off, 16));

    float s = 0.f;
    for (int m = j; m < M_DIM; m += 16)
      s += __expf(preRow[(size_t)m * HW] - mx);
    #pragma unroll
    for (int off = 8; off > 0; off >>= 1)
      s += __shfl_xor(s, off, 16);
    const float inv = 1.f / s;

    float l1 = 0.f;
    for (int m = j; m < M_DIM; m += 16) {
      float a  = __expf(preRow[(size_t)m * HW] - mx) * inv;
      float d  = a - LAMBDA;
      float a2 = d > 0.f ? d * a / (d + EPSV) : 0.f;
      l1 += a2;
    }
    #pragma unroll
    for (int off = 8; off > 0; off >>= 1)
      l1 += __shfl_xor(l1, off, 16);
    const float scale = 1.f / fmaxf(l1, EPSV);

    float* attRow = outAtt + (size_t)b * M_DIM * HW + hw0 + r;
    for (int m = j; m < M_DIM; m += 16) {
      float a   = __expf(preRow[(size_t)m * HW] - mx) * inv;
      float d   = a - LAMBDA;
      float a2  = d > 0.f ? d * a / (d + EPSV) : 0.f;
      float fin = a2 * scale;
      attRow[(size_t)m * HW] = fin;           // f32 att_o output
      attB[r * M_PAD + m] = (__bf16)fin;      // bf16 A-operand for GEMM2
    }
    attB[r * M_PAD + M_DIM + j] = (__bf16)0.0f;  // zero K-padding
  }
  __syncthreads();

  // ---- Phase 4: GEMM2  y = att @ W  (16 C-tiles, 2 per wave) --------------
  #pragma unroll
  for (int ci = 0; ci < 2; ++ci) {
    const int ct = wave * 2 + ci;
    const __bf16* ap = attB + arow * M_PAD + khalf;
    const __bf16* bp = w2 + (size_t)(ct * 16 + arow) * M_PAD + khalf;
    __builtin_prefetch(bp, 0, 1);

    // 1-deep software pipeline: preload next K-step's fragments before
    // consuming the current ones.
    v16bf af0 = *(const v16bf*)(ap);
    v16bf bf0 = *(const v16bf*)(bp);
    v8f acc = {};
    for (int k0 = 0; k0 < M_PAD; k0 += 32) {
      const int kn = (k0 + 32 < M_PAD) ? (k0 + 32) : k0;  // uniform clamp
      v16bf af1 = *(const v16bf*)(ap + kn);
      v16bf bf1 = *(const v16bf*)(bp + kn);
      acc = __builtin_amdgcn_wmma_f32_16x16x32_bf16(
          false, af0, false, bf0, (short)0, acc, false, false);
      af0 = af1;
      bf0 = bf1;
    }

    size_t p = (size_t)b * C_DIM * HW + (size_t)(ct * 16 + arow) * HW
             + (size_t)hw0 + 8 * (lane >> 4);
    #pragma unroll
    for (int v = 0; v < 8; ++v) outY[p + v] = acc[v];
  }
}

// ---------------------------------------------------------------------------
extern "C" void kernel_launch(void* const* d_in, const int* in_sizes, int n_in,
                              void* d_out, int out_size, void* d_ws, size_t ws_size,
                              hipStream_t stream) {
  const float* x = (const float*)d_in[0];   // [32,256,32,32]
  const float* w = (const float*)d_in[1];   // [2000,256]

  __bf16* w1 = (__bf16*)d_ws;               // bf16 [2000][256]
  __bf16* w2 = w1 + W1_ELEMS;               // bf16 [256][2016]

  float* outY   = (float*)d_out;                                  // [32,256,32,32]
  float* outAtt = outY   + (size_t)32 * C_DIM * HW;               // [32,2000,32,32]
  float* outPre = outAtt + (size_t)32 * M_DIM * HW;               // [32,2000,32,32]

  prep_weights<<<(W1_ELEMS + BDIM - 1) / BDIM, BDIM, 0, stream>>>(w, w1, w2);
  memae_fused<<<N_TOT / ROWS, BDIM, 0, stream>>>(x, w1, w2, outY, outAtt, outPre);
}
